// CBBIOMFP_9234179686614
// MI455X (gfx1250) — matrix operand order repaired and checked by reading, not compiled, not run on hardware
//
#include <hip/hip_runtime.h>
#include <hip/hip_bf16.h>

// ---------------------------------------------------------------------------
// MI455X (gfx1250, wave32) implementation.
// All GEMMs use v_wmma_f32_16x16x32_f16 (f16 inputs, f32 accum).
// GEMM shapes in this net satisfy M%32==0, N%64==0, K%64==0 -> guard-free,
// fully vectorized staging (global float4 -> regs -> LDS half4), software
// pipelined one K-chunk ahead with ping-pong LDS (ONE barrier per chunk)
// + global_prefetch one more chunk ahead.
// ---------------------------------------------------------------------------

#define B_    32
#define S_    512
#define D_    128
#define H_    2
#define DK_   64
#define DFF_  512
#define BS_   (B_ * S_)

typedef __attribute__((ext_vector_type(16))) _Float16 v16h;
typedef __attribute__((ext_vector_type(8)))  float    v8f;
typedef __attribute__((ext_vector_type(4)))  _Float16 h4;

// ---------------------------------------------------------------------------
// Embedding + sinusoidal positional encoding:  x[b,s,:] = emb[tok[b,s]] + pe[s]
// ---------------------------------------------------------------------------
__global__ __launch_bounds__(128) void k_embed(const int* __restrict__ tok,
                                               const float* __restrict__ emb,
                                               float* __restrict__ x) {
  int bs = blockIdx.x;            // 0..B*S-1
  int d  = threadIdx.x;           // 0..127
  int s  = bs & (S_ - 1);
  int t  = tok[bs];
  int i2 = d & ~1;
  float div = expf((float)i2 * (-9.210340371976184f / (float)D_));
  float ang = (float)s * div;
  float pe  = (d & 1) ? cosf(ang) : sinf(ang);
  x[(long)bs * D_ + d] = emb[t * D_ + d] + pe;
}

// ---------------------------------------------------------------------------
// Generic WMMA GEMM:  C = scale * (A @ B) (+epilogue)
//   block: 256 threads = 8 waves arranged 2(M) x 4(N); block tile 32x64; KC=64
//   Requires: M%32==0, N%64==0, K%64==0, lda/ldb%4==0, 16B-aligned bases.
//   A fp32 [M x K] row-major (lda); B fp32 row-major [K x N] (ldb), or if
//   transB, element (k,n) at n*ldb + k. blockIdx.z split into (z/hsplit,
//   z%hsplit) with independent strides (for [b,h] attention batching).
//   epi: 0 none | 1 bias+relu | 2 bias | 3 +residual(aux, C layout) | 4 relu
//   Ping-pong LDS: chunk i stores to buf, one barrier, WMMAs read buf while
//   chunk i+1 global loads are in flight; chunk i+1 stores go to buf^1, so no
//   trailing barrier is needed (WAR ordering comes from the previous barrier).
// ---------------------------------------------------------------------------
__global__ __launch_bounds__(256) void k_gemm(
    const float* __restrict__ A, const float* __restrict__ Bm,
    float* __restrict__ C, const float* __restrict__ aux,
    int M, int N, int K, int lda, int ldb, int ldc,
    long sAb, long sAh, long sBb, long sBh, long sCb, long sCh,
    int hsplit, int transB, int epi, float scale) {
  // rows padded to 80 halves (160 B: 16B-multiple for b128 LDS reads,
  // non-power-of-two rotation across the 64 banks); 2 ping-pong buffers.
  __shared__ _Float16 As[2][32][80];
  __shared__ _Float16 Bs[2][64][80];   // stored transposed: Bs[.][n][k]

  int zb = blockIdx.z / hsplit, zh = blockIdx.z % hsplit;
  const float* Ap = A  + zb * sAb + zh * sAh;
  const float* Bp = Bm + zb * sBb + zh * sBh;
  float*       Cp = C  + zb * sCb + zh * sCh;

  int row0 = blockIdx.y * 32;
  int col0 = blockIdx.x * 64;
  int tid  = threadIdx.x;
  int wave = tid >> 5, lane = tid & 31;
  int wy = wave >> 2, wx = wave & 3;
  int r  = lane & 15, hi = lane >> 4;

  // ---- per-thread staging coordinates & strength-reduced pointers
  int aR[2], aK[2];                 // A: (row, k) per float4
  const float* pa[2];
#pragma unroll
  for (int i = 0; i < 2; ++i) {
    int f4 = i * 256 + tid;         // 0..511
    aR[i] = f4 >> 4;                // 0..31
    aK[i] = (f4 & 15) * 4;          // 0..60
    pa[i] = Ap + (long)(row0 + aR[i]) * lda + aK[i];
  }
  int bN[4], bK[4];
  const float* pb[4];
  long bStep;                       // pointer advance per 64-K chunk
#pragma unroll
  for (int i = 0; i < 4; ++i) {
    int f4 = i * 256 + tid;         // 0..1023
    if (transB) { bN[i] = f4 >> 4;       bK[i] = (f4 & 15) * 4;
                  pb[i] = Bp + (long)(col0 + bN[i]) * ldb + bK[i]; }
    else        { bK[i] = f4 >> 4;       bN[i] = (f4 & 15) * 4;
                  pb[i] = Bp + (long)bK[i] * ldb + (col0 + bN[i]); }
  }
  bStep = transB ? 64 : (long)64 * ldb;

  // ---- prologue: load chunk 0 into registers
  float4 ra[2], rb[4];
#pragma unroll
  for (int i = 0; i < 2; ++i) { ra[i] = *(const float4*)pa[i]; pa[i] += 64; }
#pragma unroll
  for (int i = 0; i < 4; ++i) { rb[i] = *(const float4*)pb[i]; pb[i] += bStep; }

  v8f acc = {};
  int buf = 0;
  for (int k0 = 0; k0 < K; k0 += 64, buf ^= 1) {
    // ---- store current chunk (registers) to LDS[buf] as f16
#pragma unroll
    for (int i = 0; i < 2; ++i) {
      h4 hv = { (_Float16)ra[i].x, (_Float16)ra[i].y,
                (_Float16)ra[i].z, (_Float16)ra[i].w };
      *(h4*)&As[buf][aR[i]][aK[i]] = hv;
    }
    if (transB) {
#pragma unroll
      for (int i = 0; i < 4; ++i) {
        h4 hv = { (_Float16)rb[i].x, (_Float16)rb[i].y,
                  (_Float16)rb[i].z, (_Float16)rb[i].w };
        *(h4*)&Bs[buf][bN[i]][bK[i]] = hv;
      }
    } else {
#pragma unroll
      for (int i = 0; i < 4; ++i) {
        Bs[buf][bN[i] + 0][bK[i]] = (_Float16)rb[i].x;
        Bs[buf][bN[i] + 1][bK[i]] = (_Float16)rb[i].y;
        Bs[buf][bN[i] + 2][bK[i]] = (_Float16)rb[i].z;
        Bs[buf][bN[i] + 3][bK[i]] = (_Float16)rb[i].w;
      }
    }
    __syncthreads();                 // single barrier per chunk (ping-pong)

    // ---- issue global loads for chunk k0+64 (consumed next iteration)
    if (k0 + 64 < K) {
#pragma unroll
      for (int i = 0; i < 2; ++i) { ra[i] = *(const float4*)pa[i]; pa[i] += 64; }
#pragma unroll
      for (int i = 0; i < 4; ++i) { rb[i] = *(const float4*)pb[i]; pb[i] += bStep; }
      if (k0 + 128 < K) {            // keep L2 one more chunk ahead
#pragma unroll
        for (int i = 0; i < 2; ++i) __builtin_prefetch((const void*)pa[i], 0, 1);
#pragma unroll
        for (int i = 0; i < 4; ++i) __builtin_prefetch((const void*)pb[i], 0, 1);
      }
    }

    // ---- two 16x16x32 WMMAs over the 64-wide K chunk
#pragma unroll
    for (int kk = 0; kk < 64; kk += 32) {
      v16h af, bf;
#pragma unroll
      for (int i = 0; i < 16; ++i) {
        int ka = i + hi * 16;   // A: K index for half i
        af[i] = As[buf][wy * 16 + r][kk + ka];
        bf[i] = Bs[buf][wx * 16 + r][kk + hi * 16 + i];
      }
      acc = __builtin_amdgcn_wmma_f32_16x16x32_f16(
          false, af, false, bf, (short)0, acc, false, false);
    }
  }

  // ---- writeback: C/D f32 16x16 layout (VGPR v -> M = v + 8*hi, N = lane&15)
  const float* Rp = (epi == 3) ? (aux + zb * sCb + zh * sCh) : (const float*)0;
  int gcol = col0 + wx * 16 + r;
  float bias = (epi == 1 || epi == 2) ? aux[gcol] : 0.f;
#pragma unroll
  for (int v8 = 0; v8 < 8; ++v8) {
    int grow = row0 + wy * 16 + v8 + 8 * hi;
    float v = acc[v8] * scale + bias;
    if (epi == 3)             v += Rp[(long)grow * ldc + gcol];
    if (epi == 1 || epi == 4) v = fmaxf(v, 0.f);
    Cp[(long)grow * ldc + gcol] = v;
  }
}

// ---------------------------------------------------------------------------
// Masked softmax over last dim of att[B,H,S,S]; pad where tok[b,k]==0.
// ---------------------------------------------------------------------------
__global__ __launch_bounds__(256) void k_softmax(float* __restrict__ att,
                                                 const int* __restrict__ tok) {
  __shared__ float red[256];
  long row = blockIdx.x;                      // 0..B*H*S-1
  int b = (int)(row / (H_ * S_));
  float* p = att + row * S_;
  const int* tb = tok + (long)b * S_;
  int t = threadIdx.x;
  float vals[2];
  float m = -3.4e38f;
#pragma unroll
  for (int i = 0; i < 2; ++i) {
    int c = t + i * 256;
    float v = p[c];
    if (tb[c] == 0) v = -1e9f;
    vals[i] = v;
    m = fmaxf(m, v);
  }
  red[t] = m; __syncthreads();
  for (int s = 128; s > 0; s >>= 1) { if (t < s) red[t] = fmaxf(red[t], red[t + s]); __syncthreads(); }
  m = red[0]; __syncthreads();
  float sum = 0.f;
#pragma unroll
  for (int i = 0; i < 2; ++i) { vals[i] = expf(vals[i] - m); sum += vals[i]; }
  red[t] = sum; __syncthreads();
  for (int s = 128; s > 0; s >>= 1) { if (t < s) red[t] += red[t + s]; __syncthreads(); }
  float inv = 1.0f / red[0];
#pragma unroll
  for (int i = 0; i < 2; ++i) p[t + i * 256] = vals[i] * inv;
}

// ---------------------------------------------------------------------------
// LayerNorm over D=128 (biased variance, eps=1e-6). One block per row.
// ---------------------------------------------------------------------------
__global__ __launch_bounds__(128) void k_ln(const float* __restrict__ in,
                                            float* __restrict__ out) {
  __shared__ float red[128];
  long row = blockIdx.x;
  int d = threadIdx.x;
  float v = in[row * D_ + d];
  red[d] = v; __syncthreads();
  for (int s = 64; s > 0; s >>= 1) { if (d < s) red[d] += red[d + s]; __syncthreads(); }
  float mean = red[0] * (1.0f / D_); __syncthreads();
  float c = v - mean;
  red[d] = c * c; __syncthreads();
  for (int s = 64; s > 0; s >>= 1) { if (d < s) red[d] += red[d + s]; __syncthreads(); }
  float var = red[0] * (1.0f / D_);
  out[row * D_ + d] = c / sqrtf(var + 1e-6f);
}

// ---------------------------------------------------------------------------
// mask[b,s] = sum_{h,q} att[b,h,q,s]
// ---------------------------------------------------------------------------
__global__ void k_masksum(const float* __restrict__ att, float* __restrict__ mv) {
  int idx = blockIdx.x * blockDim.x + threadIdx.x;  // b*S + s
  if (idx >= B_ * S_) return;
  int b = idx / S_, s = idx & (S_ - 1);
  const float* base = att + (long)b * H_ * S_ * S_ + s;
  float acc = 0.f;
  for (int hq = 0; hq < H_ * S_; ++hq) acc += base[(long)hq * S_];
  mv[idx] = acc;
}

// ---------------------------------------------------------------------------
// _creat_mask: j = first index with mask==0 (none -> keep all); k=ceil(.2*j);
// drop the k smallest (stable ascending rank, index tie-break) among [0,j).
// ---------------------------------------------------------------------------
__global__ __launch_bounds__(512) void k_creatmask(const float* __restrict__ mv,
                                                   const int* __restrict__ pep,
                                                   int* __restrict__ mp) {
  __shared__ float sm[S_];
  __shared__ int sj;
  int b = blockIdx.x, i = threadIdx.x;
  sm[i] = mv[b * S_ + i];
  if (i == 0) sj = S_;
  __syncthreads();
  if (sm[i] == 0.0f) atomicMin(&sj, i);
  __syncthreads();
  int j = sj, k = 0;
  if (j < S_) k = (int)ceilf(0.2f * (float)j);   // has-zero case
  else        j = 0;                              // no zero -> keep all
  int keep = 1;
  if (i < j && k > 0) {
    float vi = sm[i];
    int rank = 0;
    for (int l = 0; l < j; ++l) {
      float vl = sm[l];
      if (vl < vi || (vl == vi && l < i)) ++rank;
    }
    keep = (rank >= k) ? 1 : 0;
  }
  mp[b * S_ + i] = keep ? pep[b * S_ + i] : 0;
}

// ---------------------------------------------------------------------------
// Host driver
// ---------------------------------------------------------------------------
extern "C" void kernel_launch(void* const* d_in, const int* in_sizes, int n_in,
                              void* d_out, int out_size, void* d_ws, size_t ws_size,
                              hipStream_t stream) {
  (void)in_sizes; (void)n_in; (void)out_size; (void)ws_size;

  const int*   pep  = (const int*)d_in[0];
  const float* emb  = (const float*)d_in[1];
  const float* a_w[6]; const float* t_w[6]; const float* e_w[6];
  for (int i = 0; i < 6; ++i) a_w[i] = (const float*)d_in[2 + i];
  for (int i = 0; i < 6; ++i) t_w[i] = (const float*)d_in[8 + i];
  for (int i = 0; i < 6; ++i) e_w[i] = (const float*)d_in[14 + i];
  const float* p1w = (const float*)d_in[20]; const float* p1b = (const float*)d_in[21];
  const float* p2w = (const float*)d_in[22]; const float* p2b = (const float*)d_in[23];
  const float* p3w = (const float*)d_in[24]; const float* p3b = (const float*)d_in[25];
  float* out = (float*)d_out;

  // ---- workspace layout (floats); total ~35.8M floats (~144 MB) ----
  float* W = (float*)d_ws;
  const long XSZ = (long)BS_ * D_;                 // 2,097,152
  float* X    = W;
  float* XM   = X   + XSZ;
  float* XW   = XM  + XSZ;
  float* Qb   = XW  + XSZ;
  float* Kb   = Qb  + XSZ;
  float* Vb   = Kb  + XSZ;
  float* CTX  = Vb  + XSZ;
  float* ATT  = CTX + XSZ;                         // B*H*S*S = 16,777,216
  float* FF1  = ATT;                               // aliased (disjoint lifetime)
  float* HCAT = ATT + (long)B_ * H_ * S_ * S_;     // 64 x 65536
  float* HB1  = HCAT + (long)64 * S_ * D_;
  float* HB2  = HB1 + 64 * 1024;
  float* HZ   = HB2 + 64 * 512;
  float* MASKV = HZ + 64 * D_;
  int*   MASKP = (int*)(MASKV + BS_);

  // ---- output offsets (floats) ----
  const long OH1 = 0;
  const long OZ1 = OH1 + (long)B_ * S_ * D_;
  const long OH2 = OZ1 + (long)B_ * D_;
  const long OZ2 = OH2 + (long)B_ * S_ * D_;
  const long OAT = OZ2 + (long)B_ * D_;

  auto gemm = [&](const float* A, const float* Bm, float* C, const float* aux,
                  int M, int N, int K, int lda, int ldb, int ldc,
                  long sAb, long sAh, long sBb, long sBh, long sCb, long sCh,
                  int zcount, int hsplit, int transB, int epi, float scale) {
    dim3 g((N + 63) / 64, (M + 31) / 32, zcount);
    k_gemm<<<g, 256, 0, stream>>>(A, Bm, C, aux, M, N, K, lda, ldb, ldc,
                                  sAb, sAh, sBb, sBh, sCb, sCh,
                                  hsplit, transB, epi, scale);
  };

  const size_t XBYTES = (size_t)XSZ * sizeof(float);

  auto encoder = [&](const int* tokp, const float* const wset[6], int L,
                     bool maskSum, bool lastAttnOut) {
    for (int l = 0; l < L; ++l) {
      const float* wq = wset[0] + (long)l * D_ * D_;
      const float* wk = wset[1] + (long)l * D_ * D_;
      const float* wv = wset[2] + (long)l * D_ * D_;
      const float* wo = wset[3] + (long)l * D_ * D_;
      const float* f1 = wset[4] + (long)l * D_ * DFF_;
      const float* f2 = wset[5] + (long)l * DFF_ * D_;
      // Q,K,V = XW @ w*
      gemm(XW, wq, Qb, 0, BS_, D_, D_, D_, D_, D_, 0,0,0,0,0,0, 1,1, 0, 0, 1.f);
      gemm(XW, wk, Kb, 0, BS_, D_, D_, D_, D_, D_, 0,0,0,0,0,0, 1,1, 0, 0, 1.f);
      gemm(XW, wv, Vb, 0, BS_, D_, D_, D_, D_, D_, 0,0,0,0,0,0, 1,1, 0, 0, 1.f);
      // scores[b,h] = Q_bh @ K_bh^T / sqrt(dk)
      gemm(Qb, Kb, ATT, 0, S_, S_, DK_, D_, D_, S_,
           (long)S_ * D_, DK_, (long)S_ * D_, DK_,
           (long)H_ * S_ * S_, (long)S_ * S_,
           B_ * H_, H_, 1, 0, 0.125f);
      k_softmax<<<B_ * H_ * S_, 256, 0, stream>>>(ATT, tokp);
      if (maskSum && l == 0)
        k_masksum<<<(BS_ + 255) / 256, 256, 0, stream>>>(ATT, MASKV);
      if (lastAttnOut && l == L - 1)
        hipMemcpyAsync(out + OAT, ATT, (size_t)B_ * H_ * S_ * S_ * sizeof(float),
                       hipMemcpyDeviceToDevice, stream);
      // ctx[b,h] = attn @ V_bh
      gemm(ATT, Vb, CTX, 0, S_, DK_, S_, S_, D_, D_,
           (long)H_ * S_ * S_, (long)S_ * S_, (long)S_ * D_, DK_,
           (long)S_ * D_, DK_, B_ * H_, H_, 0, 0, 1.f);
      // x = LN(ctx @ wo + x)
      gemm(CTX, wo, Qb, XW, BS_, D_, D_, D_, D_, D_, 0,0,0,0,0,0, 1,1, 0, 3, 1.f);
      k_ln<<<BS_, 128, 0, stream>>>(Qb, XW);
      // x = LN(relu(x @ f1) @ f2 + x)
      gemm(XW, f1, FF1, 0, BS_, DFF_, D_, D_, DFF_, DFF_, 0,0,0,0,0,0, 1,1, 0, 4, 1.f);
      gemm(FF1, f2, Qb, XW, BS_, D_, DFF_, DFF_, D_, D_, 0,0,0,0,0,0, 1,1, 0, 3, 1.f);
      k_ln<<<BS_, 128, 0, stream>>>(Qb, XW);
    }
  };

  // 1) x = emb[peptide] + pe
  k_embed<<<BS_, 128, 0, stream>>>(pep, emb, X);

  // 2) a-encoder (1 layer) -> attention mass -> mask
  hipMemcpyAsync(XW, X, XBYTES, hipMemcpyDeviceToDevice, stream);
  encoder(pep, a_w, 1, true, false);
  k_creatmask<<<B_, 512, 0, stream>>>(MASKV, pep, MASKP);
  k_embed<<<BS_, 128, 0, stream>>>(MASKP, emb, XM);

  // 3) e-encoder (3 layers) on original x; last-layer attn is an output
  hipMemcpyAsync(XW, X, XBYTES, hipMemcpyDeviceToDevice, stream);
  encoder(pep, e_w, 3, false, true);
  hipMemcpyAsync(out + OH1, XW, XBYTES, hipMemcpyDeviceToDevice, stream);
  hipMemcpyAsync(HCAT, XW, XBYTES, hipMemcpyDeviceToDevice, stream);

  // 4) t-encoder (2 layers) on masked embedding, masked tokens for padding
  hipMemcpyAsync(XW, XM, XBYTES, hipMemcpyDeviceToDevice, stream);
  encoder(MASKP, t_w, 2, false, false);
  hipMemcpyAsync(out + OH2, XW, XBYTES, hipMemcpyDeviceToDevice, stream);
  hipMemcpyAsync(HCAT + XSZ, XW, XBYTES, hipMemcpyDeviceToDevice, stream);

  // 5) head MLP on stacked [h1; h2] (streams 268MB p1_w once for both)
  gemm(HCAT, p1w, HB1, p1b, 64, 1024, S_ * D_, S_ * D_, 1024, 1024,
       0,0,0,0,0,0, 1,1, 0, 1, 1.f);
  gemm(HB1, p2w, HB2, p2b, 64, 512, 1024, 1024, 512, 512,
       0,0,0,0,0,0, 1,1, 0, 1, 1.f);
  gemm(HB2, p3w, HZ, p3b, 64, 128, 512, 512, 128, 128,
       0,0,0,0,0,0, 1,1, 0, 2, 1.f);
  hipMemcpyAsync(out + OZ1, HZ, (size_t)B_ * D_ * sizeof(float),
                 hipMemcpyDeviceToDevice, stream);
  hipMemcpyAsync(out + OZ2, HZ + (long)B_ * D_, (size_t)B_ * D_ * sizeof(float),
                 hipMemcpyDeviceToDevice, stream);
}
